// TopKTiledSoftmax_68968584839723
// MI455X (gfx1250) — matrix operand order repaired and checked by reading, not compiled, not run on hardware
//
#include <hip/hip_runtime.h>
#include <hip/hip_bf16.h>

// Problem constants (from reference)
#define N_ROWS   4096
#define D_DIM    1024
#define V_VOCAB  256000
#define K_TOP    128
#define T_TILES  16
#define TILE_V   (V_VOCAB / T_TILES)   // 16000
#define BM       128                   // rows of x per workgroup (LDS-resident)

typedef __attribute__((ext_vector_type(16))) __bf16 v16bf;
typedef __attribute__((ext_vector_type(8)))  float  v8f;
typedef __attribute__((ext_vector_type(4)))  unsigned int v4u;
typedef __attribute__((ext_vector_type(8)))  int  v8i;
typedef __attribute__((ext_vector_type(4)))  int  v4i;

union FragBF16 { v16bf v; uint4 u[2]; };

// ---------------------------------------------------------------------------
// Kernel 1: f32 -> bf16 conversion (grid-stride, 4 elements / iteration)
// ---------------------------------------------------------------------------
__global__ void convert_bf16_kernel(const float* __restrict__ src,
                                    __bf16* __restrict__ dst,
                                    long long n4) {
    long long i = (long long)blockIdx.x * blockDim.x + threadIdx.x;
    long long stride = (long long)gridDim.x * blockDim.x;
    for (; i < n4; i += stride) {
        float4 f = ((const float4*)src)[i];
        union { __bf16 h[4]; uint2 u; } o;
        o.h[0] = (__bf16)f.x;
        o.h[1] = (__bf16)f.y;
        o.h[2] = (__bf16)f.z;
        o.h[3] = (__bf16)f.w;
        *(uint2*)(dst + 4 * i) = o.u;
    }
}

// ---------------------------------------------------------------------------
// TDM iterate-mode load: stage 128 x 1024 bf16 rows of x (256 KB) into LDS,
// row-interleaved so LDS row = (row%16)*8 + (row/16).
// Per CDNA5 ISA 8.3-8.6 + 9.3: per iteration copy an 8-row tile at global
// row-stride 16; 16 iterations, global_addr += 1 row, lds_addr += 8 rows.
// Toolchain builtin (6 args):
//   (uint32x4 g0, int32x8 g1, int32x4 g2, int32x4 g3, int32x8 pad, i32 cpol)
// ---------------------------------------------------------------------------
__device__ __forceinline__ void tdm_load_x_tile(const __bf16* gsrc,
                                                __bf16* lds_dst) {
    unsigned long long ga = (unsigned long long)(uintptr_t)gsrc;
    unsigned lds = (unsigned)(uintptr_t)lds_dst;   // LDS aperture: addr[31:0]

    v4u g0;
    g0[0] = 1u;                                    // count=1 (valid user D#)
    g0[1] = lds;                                   // lds_addr (bytes)
    g0[2] = (unsigned)(ga & 0xFFFFFFFFu);          // global_addr[31:0]
    g0[3] = (unsigned)((ga >> 32) & 0x01FFFFFFu)   // global_addr[56:32]
          | 0x80000000u;                           // type=2 ("image")

    v8i g1;
    g1[0] = 0x00090000;            // data_size=1 (2B) | iterate_enable (b19)
    g1[1] = (int)(1024u << 16);    // tensor_dim0 = 1024      @ bits 79:48
    g1[2] = (int)(8u << 16);       // tensor_dim1 = 8         @ bits 111:80
    g1[3] = (int)(1024u << 16);    // tile_dim0  = 1024       @ bits 127:112
    g1[4] = 8;                     // tile_dim1  = 8 rows     @ bits 143:128
    g1[5] = 16384;                 // tensor_dim0_stride = 16 rows (elements)
    g1[6] = 0;
    g1[7] = 0;

    v4i g2;
    g2[0] = 0;                     // tensor_dim2 (unused)
    g2[1] = 8192;                  // lds_addr_increment    = 8 rows (elems)
    g2[2] = 1024;                  // global_addr_increment = 1 row  (elems)
    g2[3] = (int)(15u << 16);      // iterate_count = 15 (16 iterations)

    v4i z4 = {0, 0, 0, 0};
    v8i z8 = {0, 0, 0, 0, 0, 0, 0, 0};
    __builtin_amdgcn_tensor_load_to_lds(g0, g1, g2, z4, z8, 0);
}

// ---------------------------------------------------------------------------
// Kernel 2: bf16 WMMA GEMM (128 rows x vocab tile) + tournament selection.
// grid = (N/128, T), block = 256 (8 waves), dynamic LDS = 256 KB.
// Waves split columns (16 each); one B fragment feeds 8 WMMAs (all 8 row
// tiles). Interleaved LDS layout keeps all A ds_loads at immediate offsets
// from one loop-invariant base. Selection state = one packed u32 key/slot.
// ---------------------------------------------------------------------------
__global__ void __launch_bounds__(256)
gemm_topk_kernel(const __bf16* __restrict__ Xb,   // [N, D]
                 const __bf16* __restrict__ Wb,   // [V, D]
                 float* __restrict__ cand_v,      // [N, T, 128]
                 int*   __restrict__ cand_i) {    // [N, T, 128]
    extern __shared__ __bf16 xs[];                // 256 KB, row-interleaved

    const int tid  = threadIdx.x;
    const int wave = tid >> 5;
    const int lane = tid & 31;
    const int rowBlock = blockIdx.x;
    const int tile     = blockIdx.y;
    const long long tileBase = (long long)tile * TILE_V;

    if (wave == 0) {
        tdm_load_x_tile(Xb + (size_t)rowBlock * BM * D_DIM, xs);
        __builtin_amdgcn_s_wait_tensorcnt(0);
    }
    __syncthreads();

    // A-fragment (16-bit A 16x32 layout). Interleaved LDS: logical x row
    // (r*16 + arow) lives at LDS row (arow*8 + r) -> rowtile stride 2 KB.
    const int arow = lane & 15;
    const int koff = (lane >> 4) * 16;
    const __bf16* abase = xs + (size_t)arow * 8 * D_DIM + koff;
    // B-fragment (16-bit B 32x16 layout): 16 contiguous K per lane.
    const int bcol  = lane & 15;
    const int kbase = (lane >> 4) * 16;

    // Packed selection keys: [31:7] monotonic float bits, [6:0] cb iteration.
    unsigned kkey[8][8];
#pragma unroll
    for (int r = 0; r < 8; ++r)
#pragma unroll
        for (int j = 0; j < 8; ++j) kkey[r][j] = 0u;

    unsigned it = 0;
    for (int cb = wave * 16; cb < TILE_V; cb += 128, ++it) {
        const long long gcol = tileBase + cb + bcol;
        const __bf16* bptr = Wb + (size_t)gcol * D_DIM + kbase;

        // Speculative prefetch of next column block's B stream.
        if (cb + 128 < TILE_V)
            __builtin_prefetch(bptr + (size_t)128 * D_DIM, 0, 0);

        v8f acc[8];
#pragma unroll
        for (int r = 0; r < 8; ++r) acc[r] = (v8f){};

        // Software-pipelined B: b0 = current k-slice, b1 = next.
        FragBF16 b0, b1;
        {
            const uint4* bp = (const uint4*)(bptr);
            b0.u[0] = bp[0]; b0.u[1] = bp[1];
        }
#pragma unroll 2
        for (int k = 0; k < D_DIM; k += 32) {
            const int knext = (k + 32) & (D_DIM - 1);     // wrap on last iter
            const uint4* bp = (const uint4*)(bptr + knext);
            b1.u[0] = bp[0]; b1.u[1] = bp[1];

            FragBF16 a[8];
#pragma unroll
            for (int r = 0; r < 8; ++r) {
                const __bf16* ap = abase + r * D_DIM + k;
                a[r].u[0] = *(const uint4*)(ap);
                a[r].u[1] = *(const uint4*)(ap + 8);
            }
#pragma unroll
            for (int r = 0; r < 8; ++r)
                acc[r] = __builtin_amdgcn_wmma_f32_16x16x32_bf16(
                    false, a[r].v, false, b0.v, (short)0, acc[r], false, false);
            b0 = b1;
        }

        // Tournament update: one v_max_u32 per (rowtile, slot).
#pragma unroll
        for (int r = 0; r < 8; ++r)
#pragma unroll
            for (int j = 0; j < 8; ++j) {
                unsigned u = __float_as_uint(acc[r][j]);
                unsigned sgn = (unsigned)((int)u >> 31);
                unsigned mk = u ^ (sgn | 0x80000000u);     // monotonic key
                unsigned key = (mk & 0xFFFFFF80u) | it;
                kkey[r][j] = (key > kkey[r][j]) ? key : kkey[r][j];
            }
    }

    // Emit 128 candidates per row per tile: slot s = wave*16 + bcol.
    const int s = wave * 16 + bcol;
    const int rhalf = (lane >> 4) << 3;           // C/D layout: +8 rows hi half
#pragma unroll
    for (int r = 0; r < 8; ++r)
#pragma unroll
        for (int j = 0; j < 8; ++j) {
            const unsigned key = kkey[r][j];
            const unsigned itw = key & 127u;
            const unsigned mk  = key & 0xFFFFFF80u;
            const unsigned bits = (mk & 0x80000000u) ? (mk ^ 0x80000000u) : ~mk;
            const int gidx = (int)(tileBase + wave * 16 + (int)itw * 128 + bcol);
            const int m = rowBlock * BM + r * 16 + j + rhalf;
            const size_t o = ((size_t)m * T_TILES + tile) * K_TOP + s;
            cand_v[o] = __uint_as_float(bits);
            cand_i[o] = gidx;
        }
}

// ---------------------------------------------------------------------------
// Kernel 3: merge 2048 candidates/row -> top-128, softmax, gather indices,
// append target. grid = N, block = 128.
// ---------------------------------------------------------------------------
__global__ void __launch_bounds__(128)
merge_topk_softmax_kernel(const float* __restrict__ cand_v,
                          const int*   __restrict__ cand_i,
                          const long long* __restrict__ target,
                          float* __restrict__ out) {
    __shared__ float cv[T_TILES * K_TOP];
    __shared__ int   ci[T_TILES * K_TOP];
    __shared__ float redv[128];
    __shared__ int   redi[128];
    __shared__ float selv[K_TOP];
    __shared__ int   seli[K_TOP];

    const int m   = blockIdx.x;
    const int tid = threadIdx.x;
    const int NC  = T_TILES * K_TOP;        // 2048

    for (int i = tid; i < NC; i += 128) {
        cv[i] = cand_v[(size_t)m * NC + i];
        ci[i] = cand_i[(size_t)m * NC + i];
    }
    __syncthreads();

    for (int round = 0; round < K_TOP; ++round) {
        float bv = -1e30f; int bi = -1;
        for (int j = tid; j < NC; j += 128) {
            float v = cv[j];
            if (v > bv) { bv = v; bi = j; }
        }
        redv[tid] = bv; redi[tid] = bi;
        __syncthreads();
        for (int off = 64; off > 0; off >>= 1) {
            if (tid < off && redv[tid + off] > redv[tid]) {
                redv[tid] = redv[tid + off];
                redi[tid] = redi[tid + off];
            }
            __syncthreads();
        }
        if (tid == 0) {
            int w = redi[0];
            selv[round] = redv[0];
            seli[round] = (w >= 0) ? ci[w] : 0;
            if (w >= 0) cv[w] = -3.0e38f;
        }
        __syncthreads();
    }

    float mx = selv[0];
    float e  = __expf(selv[tid] - mx);
    redv[tid] = e;
    __syncthreads();
    for (int off = 64; off > 0; off >>= 1) {
        if (tid < off) redv[tid] += redv[tid + off];
        __syncthreads();
    }
    float inv = 1.0f / redv[0];

    out[(size_t)m * K_TOP + tid] = e * inv;
    float* oi = out + (size_t)N_ROWS * K_TOP;
    oi[(size_t)m * (K_TOP + 1) + tid] = (float)seli[tid];
    if (tid == 0)
        oi[(size_t)m * (K_TOP + 1) + K_TOP] = (float)target[m];
}

// ---------------------------------------------------------------------------
// Launch: convert W & x to bf16 in workspace, TDM+WMMA GEMM, merge.
// Workspace: Wbf16 (500 MB) | Xbf16 (8 MB) | cand_v (32 MB) | cand_i (32 MB)
// ---------------------------------------------------------------------------
extern "C" void kernel_launch(void* const* d_in, const int* in_sizes, int n_in,
                              void* d_out, int out_size, void* d_ws, size_t ws_size,
                              hipStream_t stream) {
    const float*     x  = (const float*)d_in[0];
    const float*     W  = (const float*)d_in[1];
    const long long* tg = (const long long*)d_in[2];

    char* ws = (char*)d_ws;
    __bf16* Wb = (__bf16*)ws;
    size_t off = (size_t)V_VOCAB * D_DIM * 2;
    __bf16* Xb = (__bf16*)(ws + off);
    off += (size_t)N_ROWS * D_DIM * 2;
    float* cand_v = (float*)(ws + off);
    off += (size_t)N_ROWS * T_TILES * K_TOP * 4;
    int* cand_i = (int*)(ws + off);

    long long nW4 = (long long)V_VOCAB * D_DIM / 4;
    long long nX4 = (long long)N_ROWS * D_DIM / 4;
    convert_bf16_kernel<<<8192, 256, 0, stream>>>(W, Wb, nW4);
    convert_bf16_kernel<<<512,  256, 0, stream>>>(x, Xb, nX4);

    dim3 grid(N_ROWS / BM, T_TILES);
    size_t ldsBytes = (size_t)BM * D_DIM * sizeof(__bf16);   // 256 KB
    gemm_topk_kernel<<<grid, 256, ldsBytes, stream>>>(Xb, Wb, cand_v, cand_i);

    merge_topk_softmax_kernel<<<N_ROWS, 128, 0, stream>>>(cand_v, cand_i, tg,
                                                          (float*)d_out);
}
